// Sequence_62251255988492
// MI455X (gfx1250) — matrix operand order repaired and checked by reading, not compiled, not run on hardware
//
#include <hip/hip_runtime.h>
#include <hip/hip_bf16.h>
#include <cstddef>

// ---------------------------------------------------------------------------
// 2-layer GRU (H=64) + linear head, T serial steps, B independent rows.
// Persistent-RNN design for MI455X (gfx1250, wave32, WMMA):
//   * one workgroup (4 wave32s) owns one 16-row batch chain
//   * GRU weight B-fragments (18 x v16h) hoisted into VGPRs (loop-invariant
//     per wave): no LDS traffic / dscnt stalls for weights in the recurrence.
//     W_lin fragments stay in LDS (wave-0-only, off the critical path).
//   * states carried in registers in WMMA D-layout; f16 LDS mirrors
//     (double-buffered -> only 3 barriers/step) feed the next A-fragments
//   * all 64-K matmuls via v_wmma_f32_16x16x32_f16 (f32 accumulate)
//   * fast transcendentals: v_rcp_f32 + v_exp_f32 (no IEEE-div expansion)
// ---------------------------------------------------------------------------

typedef __attribute__((ext_vector_type(16))) _Float16 v16h;
typedef __attribute__((ext_vector_type(8)))  float    v8f;
typedef __attribute__((ext_vector_type(4)))  float    f32x4;

#define H        64
#define G        192          // 3*H gate columns
#define INF      5            // x = [out(4), u(1)]
#define OUTF     4
#define MTILE    16           // batch rows per workgroup
#define WSTRIDE  72           // padded LDS row stride (halves) -> conflict-free

struct __align__(16) Smem {
    _Float16 wh1 [G * WSTRIDE];         // W_hh1 (192x64) row-major, f16
    _Float16 wi2 [G * WSTRIDE];         // W_ih2
    _Float16 wh2 [G * WSTRIDE];         // W_hh2
    _Float16 wlin[16 * WSTRIDE];        // W_lin (4x64) padded to 16 rows
    float    wi1 [G * 8];               // W_ih1 (192x5) padded to 8 cols, f32
    float    bih1[G], bhh1[G], bih2[G], bhh2[G];
    float    blin[4];
    _Float16 h1  [2][MTILE * WSTRIDE];  // double-buffered f16 h1 mirror
    _Float16 h2  [2][MTILE * WSTRIDE];
    float    outb[MTILE * OUTF];        // out(t-1), f32
    float    ub  [MTILE];               // u(t)
};

// --- A-fragment (16x32 f16) per ISA 7.12.2: lanes 0-15 hold M=lane,
//     elements 0..7 = K 0..7, elements 8..15 = K 16..23; lanes 16-31 are the
//     K+8 halves. Rows are contiguous in LDS -> two 16B loads per fragment.
__device__ inline v16h load_fragA(const _Float16* hbuf, int m, int kt, int hi) {
    const _Float16* p = hbuf + m * WSTRIDE + kt * 32 + hi * 8;
    v16h a;
    *reinterpret_cast<f32x4*>(&a)       = *reinterpret_cast<const f32x4*>(p);
    *(reinterpret_cast<f32x4*>(&a) + 1) = *reinterpret_cast<const f32x4*>(p + 16);
    return a;
}

// --- B-fragment (32x16 f16): lane n (0-15) holds column N=n, elements 0..7 =
//     K 0..7, 8..15 = K 16..23; lanes 16-31 the K+8 halves.  B(k,n) = W[n][k], W rows
//     contiguous in LDS -> two 16B loads per lane.
__device__ inline v16h load_fragB(const _Float16* wmat, int jt, int kt, int lane) {
    int n  = lane & 15;
    int hb = lane >> 4;
    const _Float16* p = wmat + (jt * 16 + n) * WSTRIDE + kt * 32 + hb * 8;
    v16h b;
    *reinterpret_cast<f32x4*>(&b)       = *reinterpret_cast<const f32x4*>(p);
    *(reinterpret_cast<f32x4*>(&b) + 1) = *reinterpret_cast<const f32x4*>(p + 16);
    return b;
}

__device__ inline v8f wmma16(v16h a, v16h b, v8f c) {
    return __builtin_amdgcn_wmma_f32_16x16x32_f16(false, a, false, b,
                                                  (short)0, c, false, false);
}

// fast sigmoid / tanh: v_exp_f32 + v_rcp_f32, no IEEE-division expansion
__device__ inline float sigm(float x) {
    return __builtin_amdgcn_rcpf(1.0f + __expf(-x));
}
__device__ inline float tanh_fast(float x) {
    return 2.0f * __builtin_amdgcn_rcpf(1.0f + __expf(-2.0f * x)) - 1.0f;
}

__global__ __launch_bounds__(128, 1)
void gru2_persistent(const int* __restrict__ plen,
                     const float* __restrict__ input,   // (T, B, 5)
                     const float* __restrict__ Wih1, const float* __restrict__ Whh1,
                     const float* __restrict__ Bih1, const float* __restrict__ Bhh1,
                     const float* __restrict__ Wih2, const float* __restrict__ Whh2,
                     const float* __restrict__ Bih2, const float* __restrict__ Bhh2,
                     const float* __restrict__ Wlin, const float* __restrict__ Blin,
                     float* __restrict__ out,           // (B, Tp, 4)
                     int Bsz) {
    extern __shared__ __align__(16) unsigned char smem_raw[];
    Smem& S = *reinterpret_cast<Smem*>(smem_raw);

    const int tid  = threadIdx.x;
    const int lane = tid & 31;
    const int w    = tid >> 5;          // wave id 0..3
    const int lw   = lane & 15;
    const int hi   = lane >> 4;         // lane half
    const int b0   = blockIdx.x * MTILE;
    const int Tp   = *plen;

    // ---- stage weights (f32 -> f16) into LDS, zero state buffers ----------
    for (int i = tid; i < G * H; i += 128) {
        int r = i >> 6, c = i & 63;
        S.wh1[r * WSTRIDE + c] = (_Float16)Whh1[i];
        S.wi2[r * WSTRIDE + c] = (_Float16)Wih2[i];
        S.wh2[r * WSTRIDE + c] = (_Float16)Whh2[i];
    }
    for (int i = tid; i < 16 * H; i += 128) {
        int r = i >> 6, c = i & 63;
        S.wlin[r * WSTRIDE + c] = (r < OUTF) ? (_Float16)Wlin[r * H + c] : (_Float16)0.0f;
    }
    for (int i = tid; i < G * 8; i += 128) {
        int r = i >> 3, c = i & 7;
        S.wi1[i] = (c < INF) ? Wih1[r * INF + c] : 0.0f;
    }
    for (int i = tid; i < G; i += 128) {
        S.bih1[i] = Bih1[i]; S.bhh1[i] = Bhh1[i];
        S.bih2[i] = Bih2[i]; S.bhh2[i] = Bhh2[i];
    }
    for (int i = tid; i < MTILE * WSTRIDE; i += 128) {
        S.h1[0][i] = (_Float16)0.0f; S.h1[1][i] = (_Float16)0.0f;
        S.h2[0][i] = (_Float16)0.0f; S.h2[1][i] = (_Float16)0.0f;
    }
    if (tid < OUTF)           S.blin[tid] = Blin[tid];
    if (tid < MTILE * OUTF)   S.outb[tid] = 0.0f;
    if (tid < MTILE)          S.ub[tid]   = input[(size_t)(b0 + tid) * INF + (INF - 1)];
    __syncthreads();

    // ---- loop-invariant per-lane values (this wave's h-column = colh) -----
    const int colh = w * 16 + lw;
    const float bR1 = S.bih1[colh] + S.bhh1[colh];
    const float bZ1 = S.bih1[H + colh] + S.bhh1[H + colh];
    const float biN1 = S.bih1[2 * H + colh], bhN1 = S.bhh1[2 * H + colh];
    const float bR2 = S.bih2[colh] + S.bhh2[colh];
    const float bZ2 = S.bih2[H + colh] + S.bhh2[H + colh];
    const float biN2 = S.bih2[2 * H + colh], bhN2 = S.bhh2[2 * H + colh];
    float wr[INF], wz[INF], wn[INF];
#pragma unroll
    for (int k = 0; k < INF; ++k) {
        wr[k] = S.wi1[colh * 8 + k];
        wz[k] = S.wi1[(H + colh) * 8 + k];
        wn[k] = S.wi1[(2 * H + colh) * 8 + k];
    }
    const float blin_l = (lw < OUTF) ? S.blin[lw] : 0.0f;

    // ---- hoist GRU weight B-fragments into registers (loop-invariant) -----
    v16h Bh1[3][2], Bi2[3][2], Bh2[3][2];
#pragma unroll
    for (int g = 0; g < 3; ++g)
#pragma unroll
        for (int kt = 0; kt < 2; ++kt) {
            Bh1[g][kt] = load_fragB(S.wh1, g * 4 + w, kt, lane);
            Bi2[g][kt] = load_fragB(S.wi2, g * 4 + w, kt, lane);
            Bh2[g][kt] = load_fragB(S.wh2, g * 4 + w, kt, lane);
        }

    v8f h1p = {}, h2p = {};             // carried states, f32 D-layout

    for (int t = 0; t < Tp; ++t) {
        const int p = t & 1;
        const _Float16* h1r = S.h1[p ^ 1];   // state mirrors from step t-1
        const _Float16* h2r = S.h2[p ^ 1];
        _Float16* h1w = S.h1[p];
        _Float16* h2w = S.h2[p];

        // ===== layer 1 ====================================================
        // gi1 = x @ W_ih1^T  (K=5: plain FMAs, becomes the WMMA C operand)
        v8f giR, giZ, giN;
#pragma unroll
        for (int e = 0; e < 8; ++e) {
            int m = e + hi * 8;
            f32x4 o = *reinterpret_cast<const f32x4*>(&S.outb[m * OUTF]);
            float u = S.ub[m];
            giR[e] = wr[0]*o[0] + wr[1]*o[1] + wr[2]*o[2] + wr[3]*o[3] + wr[4]*u;
            giZ[e] = wz[0]*o[0] + wz[1]*o[1] + wz[2]*o[2] + wz[3]*o[3] + wz[4]*u;
            giN[e] = wn[0]*o[0] + wn[1]*o[1] + wn[2]*o[2] + wn[3]*o[3] + wn[4]*u;
        }
        v16h ha0 = load_fragA(h1r, lw, 0, hi);
        v16h ha1 = load_fragA(h1r, lw, 1, hi);
        v8f aR = giR, aZ = giZ, ghN = {};
#pragma unroll
        for (int kt = 0; kt < 2; ++kt) {
            v16h hk = kt ? ha1 : ha0;
            aR  = wmma16(hk, Bh1[0][kt], aR);
            aZ  = wmma16(hk, Bh1[1][kt], aZ);
            ghN = wmma16(hk, Bh1[2][kt], ghN);
        }
#pragma unroll
        for (int e = 0; e < 8; ++e) {
            float r = sigm(aR[e] + bR1);
            float z = sigm(aZ[e] + bZ1);
            float n = tanh_fast(giN[e] + biN1 + r * (ghN[e] + bhN1));
            h1p[e] = (1.0f - z) * n + z * h1p[e];
        }
#pragma unroll
        for (int e = 0; e < 8; ++e)
            h1w[(e + hi * 8) * WSTRIDE + colh] = (_Float16)h1p[e];
        __syncthreads();                              // new h1 visible

        // ===== layer 2 ====================================================
        v16h x0 = load_fragA(h1w, lw, 0, hi);
        v16h x1 = load_fragA(h1w, lw, 1, hi);
        v16h g0 = load_fragA(h2r, lw, 0, hi);
        v16h g1 = load_fragA(h2r, lw, 1, hi);
        v8f aR2 = {}, aZ2 = {}, giN2 = {}, ghN2 = {};
#pragma unroll
        for (int kt = 0; kt < 2; ++kt) {
            v16h xk = kt ? x1 : x0;
            v16h gk = kt ? g1 : g0;
            aR2  = wmma16(xk, Bi2[0][kt], aR2);
            aR2  = wmma16(gk, Bh2[0][kt], aR2);
            aZ2  = wmma16(xk, Bi2[1][kt], aZ2);
            aZ2  = wmma16(gk, Bh2[1][kt], aZ2);
            giN2 = wmma16(xk, Bi2[2][kt], giN2);
            ghN2 = wmma16(gk, Bh2[2][kt], ghN2);
        }
#pragma unroll
        for (int e = 0; e < 8; ++e) {
            float r = sigm(aR2[e] + bR2);
            float z = sigm(aZ2[e] + bZ2);
            float n = tanh_fast(giN2[e] + biN2 + r * (ghN2[e] + bhN2));
            h2p[e] = (1.0f - z) * n + z * h2p[e];
        }
#pragma unroll
        for (int e = 0; e < 8; ++e)
            h2w[(e + hi * 8) * WSTRIDE + colh] = (_Float16)h2p[e];
        __syncthreads();                              // new h2 visible

        // ===== head: out = h2 @ W_lin^T + b_lin (wave 0), u prefetch (wave 1)
        if (w == 0) {
            v16h q0 = load_fragA(h2w, lw, 0, hi);
            v16h q1 = load_fragA(h2w, lw, 1, hi);
            v8f o = {};
            o = wmma16(q0, load_fragB(S.wlin, 0, 0, lane), o);
            o = wmma16(q1, load_fragB(S.wlin, 0, 1, lane), o);
            if (lw < OUTF) {
#pragma unroll
                for (int e = 0; e < 8; ++e)
                    S.outb[(e + hi * 8) * OUTF + lw] = o[e] + blin_l;
            }
        }
        if (w == 1 && lane < MTILE && (t + 1) < Tp)
            S.ub[lane] = input[((size_t)(t + 1) * Bsz + b0 + lane) * INF + (INF - 1)];
        __syncthreads();                              // outb + u(t+1) visible

        if (w == 2 && lane < MTILE) {                 // out(:, t, :) -> global
            f32x4 v = *reinterpret_cast<const f32x4*>(&S.outb[lane * OUTF]);
            *reinterpret_cast<f32x4*>(out + ((size_t)(b0 + lane) * Tp + t) * OUTF) = v;
        }
    }
}

extern "C" void kernel_launch(void* const* d_in, const int* in_sizes, int n_in,
                              void* d_out, int out_size, void* d_ws, size_t ws_size,
                              hipStream_t stream) {
    const int*   plen  = (const int*)  d_in[0];
    const float* input = (const float*)d_in[1];
    const float* Wih1  = (const float*)d_in[2];
    const float* Whh1  = (const float*)d_in[3];
    const float* bih1  = (const float*)d_in[4];
    const float* bhh1  = (const float*)d_in[5];
    const float* Wih2  = (const float*)d_in[6];
    const float* Whh2  = (const float*)d_in[7];
    const float* bih2  = (const float*)d_in[8];
    const float* bhh2  = (const float*)d_in[9];
    const float* Wlin  = (const float*)d_in[10];
    const float* blin  = (const float*)d_in[11];
    float* out = (float*)d_out;

    const int T   = 1024;
    const int Bsz = in_sizes[1] / (T * INF);          // 2048
    dim3 grid(Bsz / MTILE), block(128);
    size_t smem = sizeof(Smem);
    gru2_persistent<<<grid, block, smem, stream>>>(
        plen, input, Wih1, Whh1, bih1, bhh1,
        Wih2, Whh2, bih2, bhh2, Wlin, blin, out, Bsz);
}
